// Attention_14920716386690
// MI455X (gfx1250) — matrix operand order repaired and checked by reading, not compiled, not run on hardware
//
#include <hip/hip_runtime.h>
#include <math.h>

#define NN 10000
#define NE 160000

typedef __attribute__((ext_vector_type(16))) _Float16 v16h;
typedef __attribute__((ext_vector_type(8)))  float    v8f;

__device__ __forceinline__ v8f wmma_f16(v16h a, v16h b, v8f c) {
  // D = A(16x32 f16) * B(32x16 f16) + C(16x16 f32)
  return __builtin_amdgcn_wmma_f32_16x16x32_f16(false, a, false, b, (short)0, c, false, false);
}

__device__ __forceinline__ unsigned ordkey(float f) {
  unsigned u = __float_as_uint(f);
  return (u & 0x80000000u) ? ~u : (u | 0x80000000u);
}
__device__ __forceinline__ float inv_ordkey(unsigned k) {
  unsigned u = (k & 0x80000000u) ? (k ^ 0x80000000u) : ~k;
  return __uint_as_float(u);
}

// ---- fragment builders (contiguous k order: k = 16*(lane/16) + i) ----

// B fragment (32x16): lane<16 -> col=lane, K=0..15 ; lane>=16 -> col=lane-16, K=16..31.
__device__ __forceinline__ v16h bfrag_from_row(const float* __restrict__ row) {
  v16h b;
#pragma unroll
  for (int i = 0; i < 16; ++i) b[i] = (_Float16)row[i];
  return b;
}

// A fragment (16x32): lane<16 -> M=lane, elems = K0..15; lane>=16 -> M=lane-16, K16..31
__device__ __forceinline__ v16h afrag_from_rowmajor_f32(const float* __restrict__ base, int ld) {
  const int lane = threadIdx.x & 31;
  const int r = lane & 15;
  const int khalf = (lane & 16) ? 16 : 0;
  const float* p = base + (size_t)r * ld;
  v16h a;
#pragma unroll
  for (int i = 0; i < 16; ++i) { int k = i + khalf; a[i] = (_Float16)p[k]; }
  return a;
}

__device__ __forceinline__ v16h afrag_from_lds_h(const _Float16* __restrict__ base, int ld, int kofs) {
  const int lane = threadIdx.x & 31;
  const int r = lane & 15;
  const int khalf = (lane & 16) ? 16 : 0;
  const _Float16* p = base + r * ld + kofs;
  v16h a;
#pragma unroll
  for (int i = 0; i < 16; ++i) { int k = i + khalf; a[i] = p[k]; }
  return a;
}

// ---- core primitive: fctp for 16 items -------------------------------------
// out[item,v,m] = scale * sum_{u,a} W[v,u,a] * attr[item,a] * b[item,u,m]
// Weff tile (rows = 16 u's of fixed v, cols = 16 items) via one WMMA (K=A=32),
// then VALU apply against f16 blocks in LDS. 4-way split accumulators break the
// fmac dependency chain; v-unroll 2 keeps two independent WMMA chains in flight.
template <int D, typename StoreF>
__device__ __forceinline__ void fctp_irrep(const _Float16* __restrict__ wpack,
                                           v16h battr,
                                           const _Float16* __restrict__ bsrc, int bstride, int boff,
                                           int vout, int ntu, float scale, StoreF store) {
  const int lane = threadIdx.x & 31;
  const int col = lane & 15;
  const int rbase = (lane & 16) ? 8 : 0;
  const _Float16* bcol = bsrc + col * bstride + boff;
#pragma unroll 2
  for (int v = 0; v < vout; ++v) {
    float ac0[D], ac1[D], ac2[D], ac3[D];
#pragma unroll
    for (int m = 0; m < D; ++m) { ac0[m] = 0.f; ac1[m] = 0.f; ac2[m] = 0.f; ac3[m] = 0.f; }
    for (int ut = 0; ut < ntu; ++ut) {
      v16h a = *(const v16h*)(wpack + ((size_t)(v * ntu + ut)) * 512 + lane * 16);
      if (ut + 1 < ntu)
        __builtin_prefetch((const void*)(wpack + ((size_t)(v * ntu + ut + 1)) * 512 + lane * 16), 0, 3);
      v8f c = {};
      c = wmma_f16(a, battr, c);
      const _Float16* bl = bcol + (ut * 16 + rbase) * D;
#pragma unroll
      for (int m = 0; m < D; ++m) {
        ac0[m] = fmaf(c[0], (float)bl[0 * D + m], ac0[m]);
        ac1[m] = fmaf(c[1], (float)bl[1 * D + m], ac1[m]);
        ac2[m] = fmaf(c[2], (float)bl[2 * D + m], ac2[m]);
        ac3[m] = fmaf(c[3], (float)bl[3 * D + m], ac3[m]);
        ac0[m] = fmaf(c[4], (float)bl[4 * D + m], ac0[m]);
        ac1[m] = fmaf(c[5], (float)bl[5 * D + m], ac1[m]);
        ac2[m] = fmaf(c[6], (float)bl[6 * D + m], ac2[m]);
        ac3[m] = fmaf(c[7], (float)bl[7 * D + m], ac3[m]);
      }
    }
#pragma unroll
    for (int m = 0; m < D; ++m) {
      float acc = (ac0[m] + ac1[m]) + (ac2[m] + ac3[m]);
      float t = acc + __shfl_xor(acc, 16, 32);
      if (lane < 16) store(col, v, m, t * scale);
    }
  }
}

// ---- weight packing kernels -----------------------------------------------
// A-pack: frag f = v*ntu + ut ; tile rows r -> u = ut*16+r (W is [v][u][32], a innermost)
__global__ __launch_bounds__(32) void pack_a_kernel(const float* __restrict__ W,
                                                    _Float16* __restrict__ dst, int ntu) {
  const int f = blockIdx.x, lane = threadIdx.x;
  const int v = f / ntu, ut = f % ntu;
  const int r = lane & 15;
  const int khalf = (lane & 16) ? 16 : 0;
  const float* src = W + ((size_t)v * (ntu * 16) + (ut * 16 + r)) * 32;
  _Float16* d = dst + (size_t)f * 512 + lane * 16;
#pragma unroll
  for (int i = 0; i < 16; ++i) { int k = i + khalf; d[i] = (_Float16)src[k]; }
}

// B-pack: frag f = ct*nkt + kt ; element i -> W[(kt*32 + k0 + i)*ldn + ct*16 + col]
__global__ __launch_bounds__(32) void pack_b_kernel(const float* __restrict__ W,
                                                    _Float16* __restrict__ dst,
                                                    int nkt, int nct, int ldn) {
  const int f = blockIdx.x, lane = threadIdx.x;
  const int ct = f / nkt, kt = f % nkt;
  const int col = lane & 15;
  const int k0 = (lane & 16) ? 16 : 0;
  _Float16* d = dst + (size_t)f * 512 + lane * 16;
#pragma unroll
  for (int i = 0; i < 16; ++i)
    d[i] = (_Float16)W[(size_t)(kt * 32 + k0 + i) * ldn + ct * 16 + col];
}

// ---- init ------------------------------------------------------------------
__global__ void init_kernel(float* __restrict__ den, float* __restrict__ agg,
                            unsigned* __restrict__ mkey, int n_nodes) {
  const int t = blockIdx.x * blockDim.x + threadIdx.x;
  if (t < n_nodes * 2) { den[t] = 0.f; mkey[t] = ordkey(-3.402823466e38f); }
  if (t < n_nodes * 240) agg[t] = 0.f;
}

// ---- node-level fctps: q / k_sc / k_lin0 / v_sc / v_lin0 (hoisted off edges)
struct NodePacks { const _Float16* p[5][3]; };

__global__ __launch_bounds__(32) void node_fctp_kernel(
    const float* __restrict__ x, const float* __restrict__ attr, NodePacks packs,
    float* __restrict__ out_q, float* __restrict__ out_ksc, float* __restrict__ out_kf,
    float* __restrict__ out_vsc, float* __restrict__ out_vf) {
  __shared__ _Float16 s_x[16 * 240];
  const int lane = threadIdx.x;
  const int node0 = blockIdx.x * 16;
  for (int t = lane; t < 16 * 240; t += 32) s_x[t] = (_Float16)x[(size_t)node0 * 240 + t];
  __syncthreads();
  const int col = lane & 15;
  const int k0 = (lane & 16) ? 16 : 0;
  v16h battr = bfrag_from_row(attr + (size_t)(node0 + col) * 32 + k0);
  float* outs[5] = {out_q, out_ksc, out_kf, out_vsc, out_vf};
  const int f = blockIdx.y;
  float* op = outs[f];
  const float sc0 = rsqrtf(64.f * 32.f), sc1 = rsqrtf(32.f * 32.f), sc2 = rsqrtf(16.f * 32.f);
  fctp_irrep<1>(packs.p[f][0], battr, s_x, 240, 0, 64, 4, sc0,
    [&](int c, int v, int m, float val) { op[(size_t)(node0 + c) * 240 + v] = val; });
  fctp_irrep<3>(packs.p[f][1], battr, s_x, 240, 64, 32, 2, sc1,
    [&](int c, int v, int m, float val) { op[(size_t)(node0 + c) * 240 + 64 + v * 3 + m] = val; });
  fctp_irrep<5>(packs.p[f][2], battr, s_x, 240, 160, 16, 1, sc2,
    [&](int c, int v, int m, float val) { op[(size_t)(node0 + c) * 240 + 160 + v * 5 + m] = val; });
}

// ---- edge kernel: radial MLP (WMMA) + uvu tensor product + lin fctp (WMMA) + logits
struct EdgeArgs {
  const float *attr, *sh, *basis, *q, *kf, *vf, *ksc, *vsc;
  const int *esrc, *edst;
  const float *b1k, *b1v, *Wdot0, *Wdot1, *Wdot2;
  const _Float16 *kl0, *kl1, *kl2, *vl0, *vl1, *vl2;
  const _Float16 *W1k, *W2k, *W1v, *W2v;
  float *v_edge, *logit;
};

__global__ __launch_bounds__(32) void edge_kernel(EdgeArgs ar) {
  __shared__ int s_src[16], s_dst[16];
  __shared__ float s_sh[16][16];
  __shared__ _Float16 s_tpw[16][320];
  __shared__ _Float16 s_h[16][64];
  __shared__ _Float16 s_mid[16][896];   // mid0:0..111 (d=1), mid1:112+u*3+m (u<128), mid2:496+u*5+m (u<80)
  __shared__ float s_k[16][240];
  const int lane = threadIdx.x;
  const int e0 = blockIdx.x * 16;
  if (lane < 16) s_src[lane] = ar.esrc[e0 + lane];
  else s_dst[lane - 16] = ar.edst[e0 + lane - 16];
  for (int t = lane; t < 256; t += 32)
    s_sh[t >> 4][t & 15] = ar.sh[(size_t)(e0 + (t >> 4)) * 16 + (t & 15)];
  __syncthreads();
  const int col = lane & 15;
  const int rbase = (lane & 16) ? 8 : 0;
  const int k0h = (lane & 16) ? 16 : 0;
  v16h battr = bfrag_from_row(ar.attr + (size_t)s_src[col] * 32 + k0h);
  v16h basisA = afrag_from_rowmajor_f32(ar.basis + (size_t)e0 * 32, 32);

  for (int mod = 0; mod < 2; ++mod) {
    const _Float16* W1 = mod ? ar.W1v : ar.W1k;
    const _Float16* W2 = mod ? ar.W2v : ar.W2k;
    const float* b1 = mod ? ar.b1v : ar.b1k;
    const float* fpN = mod ? ar.vf : ar.kf;
    const _Float16* L0 = mod ? ar.vl0 : ar.kl0;
    const _Float16* L1 = mod ? ar.vl1 : ar.kl1;
    const _Float16* L2 = mod ? ar.vl2 : ar.kl2;

    // radial MLP: h = silu(basis@W1 + b1) ; tpw = h@W2
    for (int t = 0; t < 4; ++t) {
      v16h bw = *(const v16h*)(W1 + (size_t)t * 512 + lane * 16);
      v8f c = {};
      c = wmma_f16(basisA, bw, c);
      const float bias = b1[t * 16 + col];
#pragma unroll
      for (int i = 0; i < 8; ++i) {
        float z = c[i] + bias;
        s_h[rbase + i][t * 16 + col] = (_Float16)(z / (1.f + __expf(-z)));
      }
    }
    __syncthreads();
    v16h hA0 = afrag_from_lds_h(&s_h[0][0], 64, 0);
    v16h hA1 = afrag_from_lds_h(&s_h[0][0], 64, 32);
    for (int ct = 0; ct < 20; ++ct) {
      v16h bw0 = *(const v16h*)(W2 + ((size_t)(ct * 2 + 0)) * 512 + lane * 16);
      v16h bw1 = *(const v16h*)(W2 + ((size_t)(ct * 2 + 1)) * 512 + lane * 16);
      v8f c = {};
      c = wmma_f16(hA0, bw0, c);
      c = wmma_f16(hA1, bw1, c);
#pragma unroll
      for (int i = 0; i < 8; ++i) s_tpw[rbase + i][ct * 16 + col] = (_Float16)c[i];
    }
    __syncthreads();

    // uvu tensor product (elementwise): 2 lanes per edge
    {
      const int e = col, half = lane >> 4;
      const float* fb = fpN + (size_t)s_src[e] * 240;
      const float y0 = s_sh[e][0];
      const float y1x = s_sh[e][1], y1y = s_sh[e][2], y1z = s_sh[e][3];
      for (int u = half; u < 64; u += 2) {
        const float su = fb[u];
        const float w0 = (float)s_tpw[e][u], w1 = (float)s_tpw[e][64 + u], w2 = (float)s_tpw[e][128 + u];
        s_mid[e][u] = (_Float16)(su * y0 * w0);
        _Float16* m1 = &s_mid[e][112 + u * 3];
        m1[0] = (_Float16)(su * y1x * w1); m1[1] = (_Float16)(su * y1y * w1); m1[2] = (_Float16)(su * y1z * w1);
        _Float16* m2 = &s_mid[e][496 + u * 5];
#pragma unroll
        for (int m = 0; m < 5; ++m) m2[m] = (_Float16)(su * s_sh[e][4 + m] * w2);
      }
      for (int u = half; u < 32; u += 2) {
        const float* vb = fb + 64 + u * 3;
        const float a0 = vb[0], a1 = vb[1], a2 = vb[2];
        const float w3 = (float)s_tpw[e][192 + u], w4 = (float)s_tpw[e][224 + u], w5 = (float)s_tpw[e][256 + u];
        _Float16* p3 = &s_mid[e][112 + (64 + u) * 3];
        p3[0] = (_Float16)(a0 * y0 * w3); p3[1] = (_Float16)(a1 * y0 * w3); p3[2] = (_Float16)(a2 * y0 * w3);
        s_mid[e][64 + u] = (_Float16)((a0 * y1x + a1 * y1y + a2 * y1z) * w4 * 0.5773502691896258f);
        const float r2c = 0.7071067811865476f;
        _Float16* p5 = &s_mid[e][112 + (96 + u) * 3];
        p5[0] = (_Float16)((a1 * y1z - a2 * y1y) * w5 * r2c);
        p5[1] = (_Float16)((a2 * y1x - a0 * y1z) * w5 * r2c);
        p5[2] = (_Float16)((a0 * y1y - a1 * y1x) * w5 * r2c);
      }
      for (int u = half; u < 16; u += 2) {
        const float* vb = fb + 160 + u * 5;
        const float w6 = (float)s_tpw[e][288 + u], w7 = (float)s_tpw[e][304 + u];
        _Float16* p6 = &s_mid[e][496 + (64 + u) * 5];
        float dot = 0.f;
#pragma unroll
        for (int m = 0; m < 5; ++m) { const float b = vb[m]; p6[m] = (_Float16)(b * y0 * w6); dot = fmaf(b, s_sh[e][4 + m], dot); }
        s_mid[e][96 + u] = (_Float16)(dot * w7 * 0.4472135954999579f);
      }
    }
    __syncthreads();

    // lin fctp on mid (M = 112/128/80), add hoisted sc[src]
    const float sA = rsqrtf(112.f * 32.f), sB = rsqrtf(128.f * 32.f), sC = rsqrtf(80.f * 32.f);
    if (mod == 0) {
      const float* scN = ar.ksc;
      fctp_irrep<1>(L0, battr, &s_mid[0][0], 896, 0, 64, 7, sA,
        [&](int c, int v, int m, float val) { s_k[c][v] = val + scN[(size_t)s_src[c] * 240 + v]; });
      fctp_irrep<3>(L1, battr, &s_mid[0][0], 896, 112, 32, 8, sB,
        [&](int c, int v, int m, float val) { s_k[c][64 + v * 3 + m] = val + scN[(size_t)s_src[c] * 240 + 64 + v * 3 + m]; });
      fctp_irrep<5>(L2, battr, &s_mid[0][0], 896, 496, 16, 5, sC,
        [&](int c, int v, int m, float val) { s_k[c][160 + v * 5 + m] = val + scN[(size_t)s_src[c] * 240 + 160 + v * 5 + m]; });
      __syncthreads();
      // logits: lane = (edge, head)
      {
        const int e = col, h = lane >> 4;
        const float* qb = ar.q + (size_t)s_dst[e] * 240;
        float lg = 0.f;
        { const float* W = ar.Wdot0 + (size_t)h * 1024;
          float t[32];
#pragma unroll
          for (int v = 0; v < 32; ++v) t[v] = 0.f;
          for (int u = 0; u < 32; ++u) { const float qu = qb[h * 32 + u]; const float* Wr = W + u * 32;
            for (int v = 0; v < 32; ++v) t[v] = fmaf(Wr[v], qu, t[v]); }
          float s = 0.f; for (int v = 0; v < 32; ++v) s = fmaf(t[v], s_k[e][h * 32 + v], s);
          lg += s * (1.f / 32.f); }
        { const float* W = ar.Wdot1 + (size_t)h * 256;
          float s = 0.f;
          for (int m = 0; m < 3; ++m) {
            float t[16]; for (int v = 0; v < 16; ++v) t[v] = 0.f;
            for (int u = 0; u < 16; ++u) { const float qu = qb[64 + (h * 16 + u) * 3 + m]; const float* Wr = W + u * 16;
              for (int v = 0; v < 16; ++v) t[v] = fmaf(Wr[v], qu, t[v]); }
            for (int v = 0; v < 16; ++v) s = fmaf(t[v], s_k[e][64 + (h * 16 + v) * 3 + m], s); }
          lg += s * (1.f / 16.f); }
        { const float* W = ar.Wdot2 + (size_t)h * 64;
          float s = 0.f;
          for (int m = 0; m < 5; ++m) {
            float t[8]; for (int v = 0; v < 8; ++v) t[v] = 0.f;
            for (int u = 0; u < 8; ++u) { const float qu = qb[160 + (h * 8 + u) * 5 + m]; const float* Wr = W + u * 8;
              for (int v = 0; v < 8; ++v) t[v] = fmaf(Wr[v], qu, t[v]); }
            for (int v = 0; v < 8; ++v) s = fmaf(t[v], s_k[e][160 + (h * 8 + v) * 5 + m], s); }
          lg += s * (1.f / 8.f); }
        ar.logit[(size_t)(e0 + e) * 2 + h] = lg;
      }
      __syncthreads();
    } else {
      const float* scN = ar.vsc;
      float* vo = ar.v_edge;
      fctp_irrep<1>(L0, battr, &s_mid[0][0], 896, 0, 64, 7, sA,
        [&](int c, int v, int m, float val) { vo[(size_t)(e0 + c) * 240 + v] = val + scN[(size_t)s_src[c] * 240 + v]; });
      fctp_irrep<3>(L1, battr, &s_mid[0][0], 896, 112, 32, 8, sB,
        [&](int c, int v, int m, float val) { vo[(size_t)(e0 + c) * 240 + 64 + v * 3 + m] = val + scN[(size_t)s_src[c] * 240 + 64 + v * 3 + m]; });
      fctp_irrep<5>(L2, battr, &s_mid[0][0], 896, 496, 16, 5, sC,
        [&](int c, int v, int m, float val) { vo[(size_t)(e0 + c) * 240 + 160 + v * 5 + m] = val + scN[(size_t)s_src[c] * 240 + 160 + v * 5 + m]; });
    }
  }
}

// ---- segment softmax + scatter --------------------------------------------
__global__ void logit_max_kernel(const float* __restrict__ logit, const int* __restrict__ edst,
                                 unsigned* __restrict__ mkey, int n) {
  const int t = blockIdx.x * blockDim.x + threadIdx.x;
  if (t >= n) return;
  const int e = t >> 1, h = t & 1;
  atomicMax(&mkey[(size_t)edst[e] * 2 + h], ordkey(logit[t]));
}

__global__ void exp_den_kernel(const float* __restrict__ logit, const int* __restrict__ edst,
                               const unsigned* __restrict__ mkey, float* __restrict__ ex,
                               float* __restrict__ den, int n) {
  const int t = blockIdx.x * blockDim.x + threadIdx.x;
  if (t >= n) return;
  const int e = t >> 1, h = t & 1;
  const float m = inv_ordkey(mkey[(size_t)edst[e] * 2 + h]);
  const float v = __expf(logit[t] - m);
  ex[t] = v;
  atomicAdd(&den[(size_t)edst[e] * 2 + h], v);
}

__global__ void alpha_kernel(const float* __restrict__ ex, const float* __restrict__ den,
                             const int* __restrict__ edst, float* __restrict__ alpha, int n) {
  const int t = blockIdx.x * blockDim.x + threadIdx.x;
  if (t >= n) return;
  const int e = t >> 1, h = t & 1;
  alpha[t] = ex[t] / (den[(size_t)edst[e] * 2 + h] + 1e-12f);
}

__global__ void agg_kernel(const float* __restrict__ alpha, const int* __restrict__ edst,
                           const float* __restrict__ v_edge, float* __restrict__ agg, int n) {
  const int t = blockIdx.x * blockDim.x + threadIdx.x;
  if (t >= n) return;
  const int e = t / 240, ch = t - e * 240;
  int h;
  if (ch < 64) h = (ch >= 32);
  else if (ch < 160) h = ((ch - 64) / 3) >= 16;
  else h = ((ch - 160) / 5) >= 8;
  atomicAdd(&agg[(size_t)edst[e] * 240 + ch], alpha[(size_t)e * 2 + h] * v_edge[t]);
}

// ---- final: y = fctp(agg,Wlin)+fctp(x,Wsc) ; LN on l=0, RMS-scale on l=1,2
struct FinalPacks { const _Float16* lin[3]; const _Float16* sc[3]; };

__global__ __launch_bounds__(32) void final_kernel(
    const float* __restrict__ agg, const float* __restrict__ x, const float* __restrict__ attr,
    FinalPacks packs, const float* __restrict__ ln_w, const float* __restrict__ ln_b,
    float* __restrict__ out) {
  __shared__ _Float16 s_a[16 * 240];
  __shared__ _Float16 s_x[16 * 240];
  __shared__ float s_y[16][240];
  const int lane = threadIdx.x;
  const int node0 = blockIdx.x * 16;
  for (int t = lane; t < 16 * 240; t += 32) {
    s_a[t] = (_Float16)agg[(size_t)node0 * 240 + t];
    s_x[t] = (_Float16)x[(size_t)node0 * 240 + t];
    s_y[t / 240][t % 240] = 0.f;
  }
  __syncthreads();
  const int col = lane & 15;
  const int k0 = (lane & 16) ? 16 : 0;
  v16h battr = bfrag_from_row(attr + (size_t)(node0 + col) * 32 + k0);
  const float sc0 = rsqrtf(64.f * 32.f), sc1 = rsqrtf(32.f * 32.f), sc2 = rsqrtf(16.f * 32.f);
  fctp_irrep<1>(packs.lin[0], battr, s_a, 240, 0, 64, 4, sc0,
    [&](int c, int v, int m, float val) { s_y[c][v] += val; });
  fctp_irrep<3>(packs.lin[1], battr, s_a, 240, 64, 32, 2, sc1,
    [&](int c, int v, int m, float val) { s_y[c][64 + v * 3 + m] += val; });
  fctp_irrep<5>(packs.lin[2], battr, s_a, 240, 160, 16, 1, sc2,
    [&](int c, int v, int m, float val) { s_y[c][160 + v * 5 + m] += val; });
  fctp_irrep<1>(packs.sc[0], battr, s_x, 240, 0, 64, 4, sc0,
    [&](int c, int v, int m, float val) { s_y[c][v] += val; });
  fctp_irrep<3>(packs.sc[1], battr, s_x, 240, 64, 32, 2, sc1,
    [&](int c, int v, int m, float val) { s_y[c][64 + v * 3 + m] += val; });
  fctp_irrep<5>(packs.sc[2], battr, s_x, 240, 160, 16, 1, sc2,
    [&](int c, int v, int m, float val) { s_y[c][160 + v * 5 + m] += val; });
  __syncthreads();
  const int node = col, half = lane >> 4;
  float* y = s_y[node];
  float* op = out + (size_t)(node0 + node) * 240;
  // layernorm on 64 scalar channels
  float sum = 0.f, sq = 0.f;
  for (int c = half * 32; c < half * 32 + 32; ++c) { const float s = y[c]; sum += s; sq = fmaf(s, s, sq); }
  sum += __shfl_xor(sum, 16, 32); sq += __shfl_xor(sq, 16, 32);
  const float mu = sum * (1.f / 64.f);
  const float var = sq * (1.f / 64.f) - mu * mu;
  const float inv = rsqrtf(var + 1e-5f);
  for (int c = half * 32; c < half * 32 + 32; ++c) op[c] = (y[c] - mu) * inv * ln_w[c] + ln_b[c];
  // l=1 RMS scale
  float fn = 0.f;
  for (int t = half * 48; t < half * 48 + 48; ++t) { const float b = y[64 + t]; fn = fmaf(b, b, fn); }
  fn += __shfl_xor(fn, 16, 32); fn *= (1.f / 96.f);
  const float r1 = rsqrtf(fn + 1e-5f);
  for (int u = half * 16; u < half * 16 + 16; ++u) {
    const float s = r1 * ln_w[64 + u];
    for (int m = 0; m < 3; ++m) op[64 + u * 3 + m] = y[64 + u * 3 + m] * s;
  }
  // l=2 RMS scale
  float fn2 = 0.f;
  for (int t = half * 40; t < half * 40 + 40; ++t) { const float b = y[160 + t]; fn2 = fmaf(b, b, fn2); }
  fn2 += __shfl_xor(fn2, 16, 32); fn2 *= (1.f / 80.f);
  const float r2v = rsqrtf(fn2 + 1e-5f);
  for (int u = half * 8; u < half * 8 + 8; ++u) {
    const float s = r2v * ln_w[96 + u];
    for (int m = 0; m < 5; ++m) op[160 + u * 5 + m] = y[160 + u * 5 + m] * s;
  }
}

// ---- host orchestration ----------------------------------------------------
extern "C" void kernel_launch(void* const* d_in, const int* in_sizes, int n_in,
                              void* d_out, int out_size, void* d_ws, size_t ws_size,
                              hipStream_t stream) {
  auto F = [&](int i) { return (const float*)d_in[i]; };
  const float* x = F(0);
  const float* attr = F(1);
  const float* sh = F(2);
  const float* basis = F(3);
  const int* esrc = (const int*)d_in[4];
  const int* edst = (const int*)d_in[5];
  // pytree order: Wq 6-8, Wk_sc 9-11, Wk_lin0 12-14, Wk_lin 15-17, fck 18-20,
  // Wv_sc 21-23, Wv_lin0 24-26, Wv_lin 27-29, fcv 30-32, Wdot 33-35,
  // Wlin 36-38, Wsc 39-41, ln_w 42, ln_b 43
  char* ws = (char*)d_ws;
  size_t off = 0;
  auto take = [&](size_t bytes) { size_t o = off; off += (bytes + 255) & ~(size_t)255; return o; };

  const int vs[3] = {64, 32, 16};
  const int us_uu[3] = {64, 32, 16};
  const int us_um[3] = {112, 128, 80};
  const int baseIdx[9] = {6, 9, 12, 15, 21, 24, 27, 36, 39};
  const bool isUM[9] = {false, false, false, true, false, false, true, false, false};
  _Float16* P[9][3];
  for (int s = 0; s < 9; ++s)
    for (int i = 0; i < 3; ++i) {
      const int u = isUM[s] ? us_um[i] : us_uu[i];
      P[s][i] = (_Float16*)(ws + take((size_t)vs[i] * (u / 16) * 1024));
    }
  _Float16* PW1k = (_Float16*)(ws + take(4 * 1024));
  _Float16* PW2k = (_Float16*)(ws + take(40 * 1024));
  _Float16* PW1v = (_Float16*)(ws + take(4 * 1024));
  _Float16* PW2v = (_Float16*)(ws + take(40 * 1024));

  float* q_nd = (float*)(ws + take((size_t)NN * 240 * 4));
  float* ksc_nd = (float*)(ws + take((size_t)NN * 240 * 4));
  float* kf_nd = (float*)(ws + take((size_t)NN * 240 * 4));
  float* vsc_nd = (float*)(ws + take((size_t)NN * 240 * 4));
  float* vf_nd = (float*)(ws + take((size_t)NN * 240 * 4));
  float* v_edge = (float*)(ws + take((size_t)NE * 240 * 4));
  float* logit = (float*)(ws + take((size_t)NE * 2 * 4));
  float* exv = (float*)(ws + take((size_t)NE * 2 * 4));
  float* alph = (float*)(ws + take((size_t)NE * 2 * 4));
  unsigned* mkey = (unsigned*)(ws + take((size_t)NN * 2 * 4));
  float* den = (float*)(ws + take((size_t)NN * 2 * 4));
  float* aggp = (float*)(ws + take((size_t)NN * 240 * 4));
  (void)ws_size; (void)in_sizes; (void)n_in; (void)out_size;

  // 1) pack weights into WMMA fragment layout (f16)
  for (int s = 0; s < 9; ++s)
    for (int i = 0; i < 3; ++i) {
      const int u = isUM[s] ? us_um[i] : us_uu[i];
      pack_a_kernel<<<dim3(vs[i] * (u / 16)), 32, 0, stream>>>(F(baseIdx[s] + i), P[s][i], u / 16);
    }
  pack_b_kernel<<<4, 32, 0, stream>>>(F(18), PW1k, 1, 4, 64);
  pack_b_kernel<<<40, 32, 0, stream>>>(F(20), PW2k, 2, 20, 320);
  pack_b_kernel<<<4, 32, 0, stream>>>(F(30), PW1v, 1, 4, 64);
  pack_b_kernel<<<40, 32, 0, stream>>>(F(32), PW2v, 2, 20, 320);

  // 2) init softmax/agg state
  init_kernel<<<(NN * 240 + 255) / 256, 256, 0, stream>>>(den, aggp, mkey, NN);

  // 3) node-level fctps (q + hoisted sc/lin0 for both modules)
  NodePacks np;
  const int nodeSets[5] = {0, 1, 2, 4, 5};
  for (int f = 0; f < 5; ++f)
    for (int i = 0; i < 3; ++i) np.p[f][i] = P[nodeSets[f]][i];
  node_fctp_kernel<<<dim3(NN / 16, 5), 32, 0, stream>>>(x, attr, np, q_nd, ksc_nd, kf_nd, vsc_nd, vf_nd);

  // 4) edge kernel
  EdgeArgs ea;
  ea.attr = attr; ea.sh = sh; ea.basis = basis;
  ea.q = q_nd; ea.kf = kf_nd; ea.vf = vf_nd; ea.ksc = ksc_nd; ea.vsc = vsc_nd;
  ea.esrc = esrc; ea.edst = edst;
  ea.b1k = F(19); ea.b1v = F(31);
  ea.Wdot0 = F(33); ea.Wdot1 = F(34); ea.Wdot2 = F(35);
  ea.kl0 = P[3][0]; ea.kl1 = P[3][1]; ea.kl2 = P[3][2];
  ea.vl0 = P[6][0]; ea.vl1 = P[6][1]; ea.vl2 = P[6][2];
  ea.W1k = PW1k; ea.W2k = PW2k; ea.W1v = PW1v; ea.W2v = PW2v;
  ea.v_edge = v_edge; ea.logit = logit;
  edge_kernel<<<NE / 16, 32, 0, stream>>>(ea);

  // 5) segment softmax + weighted scatter
  logit_max_kernel<<<(NE * 2 + 255) / 256, 256, 0, stream>>>(logit, edst, mkey, NE * 2);
  exp_den_kernel<<<(NE * 2 + 255) / 256, 256, 0, stream>>>(logit, edst, mkey, exv, den, NE * 2);
  alpha_kernel<<<(NE * 2 + 255) / 256, 256, 0, stream>>>(exv, den, edst, alph, NE * 2);
  agg_kernel<<<(NE * 240 + 255) / 256, 256, 0, stream>>>(alph, edst, v_edge, aggp, NE * 240);

  // 6) output projection + norms
  FinalPacks fp;
  for (int i = 0; i < 3; ++i) { fp.lin[i] = P[7][i]; fp.sc[i] = P[8][i]; }
  final_kernel<<<NN / 16, 32, 0, stream>>>(aggp, x, attr, fp, F(42), F(43), (float*)d_out);
}
